// NTXent_64226940944679
// MI455X (gfx1250) — matrix operand order repaired and checked by reading, not compiled, not run on hardware
//
#include <hip/hip_runtime.h>
#include <hip/hip_bf16.h>

// ---------------------------------------------------------------------------
// NT-Xent fused loss for MI455X (gfx1250, wave32, WMMA, 320KB-LDS blocking)
//   N = 8192 rows, K = 768, tau = 0.5 -> logits = 2*dot in [-2,2]
//   Fixed-shift streaming logsumexp (shift = 2) fused into WMMA GEMM epilogue.
//   GEMM: A panel (128 x 768 f16, 194KB) + B panel (64 x 768 f16, 97KB) live
//   in LDS (padded stride -> conflict-free ds_load_b128); inner loop is pure
//   ds_load + v_wmma, no global traffic.
// ---------------------------------------------------------------------------

typedef _Float16 half8 __attribute__((ext_vector_type(8)));
typedef _Float16 v16h  __attribute__((ext_vector_type(16)));
typedef float    v8f   __attribute__((ext_vector_type(8)));

static constexpr int DIMK   = 768;     // feature dim (K)
static constexpr int NROWS  = 8192;    // 2*B
static constexpr int HALFN  = 4096;    // B (power of two: pair = row ^ HALFN)
static constexpr float TAU_INV = 2.0f; // 1/tau
static constexpr float SHIFT   = 2.0f; // max possible logit (|dot|<=1)

static constexpr int BM   = 128;       // rows per workgroup (A panel)
static constexpr int BN   = 64;        // cols per staged B panel (ct group)
static constexpr int CB   = 2048;      // cols per workgroup (CB/BN ct groups)
static constexpr int LDAB = DIMK + 8;  // padded LDS row stride in halves
                                       // (388 dwords -> conflict-free)
static constexpr int SMEM_BYTES = (BM + BN) * LDAB * 2; // 297,984 B <= 320KB

// -------------------------------------------------------------- normalize ---
// One wave per row: y[row,:] = x[row,:]/||x[row,:]|| stored as f16.
// (Reference's two-stage normalize collapses analytically to x/||x||; the
//  eps clamps are inactive since ||x_hat|| ~= 1.)  Also zeroes S[row].
__global__ void __launch_bounds__(256)
ntx_nrm(const float* __restrict__ xi, const float* __restrict__ xj,
        _Float16* __restrict__ y, float* __restrict__ S) {
    const int row  = blockIdx.x * 8 + (threadIdx.x >> 5);
    const int lane = threadIdx.x & 31;
    const float* src = (row < HALFN) ? (xi + (size_t)row * DIMK)
                                     : (xj + (size_t)(row - HALFN) * DIMK);
    float vals[DIMK / 32];
    float ss = 0.f;
#pragma unroll
    for (int i = 0; i < DIMK / 32; ++i) {
        float v = src[lane + 32 * i];
        vals[i] = v;
        ss += v * v;
    }
#pragma unroll
    for (int m = 1; m < 32; m <<= 1) ss += __shfl_xor(ss, m, 32);
    const float norm = sqrtf(ss);
    const float inv1 = 1.f / (norm + 1e-10f);          // first normalize
    const float n    = norm * inv1;                    // norm of x_hat (~1)
    const float scl  = inv1 / fmaxf(n, 1e-20f);        // second normalize
    _Float16* dst = y + (size_t)row * DIMK;
#pragma unroll
    for (int i = 0; i < DIMK / 32; ++i)
        dst[lane + 32 * i] = (_Float16)(vals[i] * scl);
    if (lane == 0) S[row] = 0.f;
}

// ---------------------------------------- LDS-blocked WMMA GEMM + fused LSE
// Grid (64, 4): blockIdx.x -> 128-row block, blockIdx.y -> 2048-col chunk.
// Block = 128 threads = 4 waves; wave w owns rows [w*32, w*32+32) of the A
// panel and computes 32x64 register tiles (2 A frags x 4 B frags = 8 WMMA
// per K-step, all operands from LDS).
// acc layout (f32 16x16 C/D): lane l, vgpr v -> row = tile_r0 + v + 8*(l>>4),
//                                               col = tile_c0 + (l&15).
__global__ void __launch_bounds__(128)
ntx_simlse(const _Float16* __restrict__ y, float* __restrict__ S) {
    extern __shared__ __align__(16) char smem[];
    _Float16* sA = (_Float16*)smem;            // [BM][LDAB]
    _Float16* sB = sA + BM * LDAB;             // [BN][LDAB]

    const int tid  = threadIdx.x;
    const int lane = tid & 31;
    const int wave = tid >> 5;
    const int l15  = lane & 15;
    const int lh   = lane >> 4;

    const int rBase = blockIdx.x * BM;
    const int c0    = blockIdx.y * CB;

    // ---- stage the 128-row A panel (contiguous in y) into LDS, padded ----
    {
        const float4* gA = (const float4*)(y + (size_t)rBase * DIMK);
        for (int i = tid; i < BM * (DIMK / 8); i += 128) {
            const int row = i / (DIMK / 8);
            const int c   = i - row * (DIMK / 8);
            *(float4*)(sA + row * LDAB + c * 8) = gA[i];
        }
    }

    // per-lane fragment base addresses in LDS
    const _Float16* a0p = sA + (size_t)(wave * 32 + l15) * LDAB + lh * 8;
    const _Float16* a1p = a0p + 16 * LDAB;

    float sumacc[2][8];
#pragma unroll
    for (int r = 0; r < 2; ++r)
#pragma unroll
        for (int v = 0; v < 8; ++v) sumacc[r][v] = 0.f;

    for (int ct = 0; ct < CB / BN; ++ct) {
        __syncthreads();   // A staged (first iter) / previous B reads done
        // ---- stage the 64-col B panel (rows of y, contiguous) ----
        {
            const float4* gB =
                (const float4*)(y + (size_t)(c0 + ct * BN) * DIMK);
            for (int i = tid; i < BN * (DIMK / 8); i += 128) {
                const int row = i / (DIMK / 8);
                const int c   = i - row * (DIMK / 8);
                *(float4*)(sB + row * LDAB + c * 8) = gB[i];
            }
        }
        __syncthreads();

        v8f acc[8];        // [r*4 + t]
#pragma unroll
        for (int t = 0; t < 8; ++t)
#pragma unroll
            for (int v = 0; v < 8; ++v) acc[t][v] = 0.f;

        const _Float16* bp0 = sB + (size_t)(l15)      * LDAB + lh * 8;
        const _Float16* bp1 = bp0 + 16 * LDAB;
        const _Float16* bp2 = bp0 + 32 * LDAB;
        const _Float16* bp3 = bp0 + 48 * LDAB;

        for (int k = 0; k < DIMK; k += 32) {
            // A fragments: K halves {lh*8..+7} and {lh*8+16..+23}
            half8 a00 = *(const half8*)(a0p + k);
            half8 a01 = *(const half8*)(a0p + k + 16);
            half8 a10 = *(const half8*)(a1p + k);
            half8 a11 = *(const half8*)(a1p + k + 16);
            v16h A0 = __builtin_shufflevector(a00, a01, 0, 1, 2, 3, 4, 5, 6, 7,
                                              8, 9, 10, 11, 12, 13, 14, 15);
            v16h A1 = __builtin_shufflevector(a10, a11, 0, 1, 2, 3, 4, 5, 6, 7,
                                              8, 9, 10, 11, 12, 13, 14, 15);
            // B fragments: K halves {lh*8..+7} and {lh*8+16..+23}
            half8 b00 = *(const half8*)(bp0 + k);
            half8 b01 = *(const half8*)(bp0 + k + 16);
            half8 b10 = *(const half8*)(bp1 + k);
            half8 b11 = *(const half8*)(bp1 + k + 16);
            half8 b20 = *(const half8*)(bp2 + k);
            half8 b21 = *(const half8*)(bp2 + k + 16);
            half8 b30 = *(const half8*)(bp3 + k);
            half8 b31 = *(const half8*)(bp3 + k + 16);
            v16h B0 = __builtin_shufflevector(b00, b01, 0, 1, 2, 3, 4, 5, 6, 7,
                                              8, 9, 10, 11, 12, 13, 14, 15);
            v16h B1 = __builtin_shufflevector(b10, b11, 0, 1, 2, 3, 4, 5, 6, 7,
                                              8, 9, 10, 11, 12, 13, 14, 15);
            v16h B2 = __builtin_shufflevector(b20, b21, 0, 1, 2, 3, 4, 5, 6, 7,
                                              8, 9, 10, 11, 12, 13, 14, 15);
            v16h B3 = __builtin_shufflevector(b30, b31, 0, 1, 2, 3, 4, 5, 6, 7,
                                              8, 9, 10, 11, 12, 13, 14, 15);

            acc[0] = __builtin_amdgcn_wmma_f32_16x16x32_f16(
                false, A0, false, B0, (short)0, acc[0], false, false);
            acc[1] = __builtin_amdgcn_wmma_f32_16x16x32_f16(
                false, A0, false, B1, (short)0, acc[1], false, false);
            acc[2] = __builtin_amdgcn_wmma_f32_16x16x32_f16(
                false, A0, false, B2, (short)0, acc[2], false, false);
            acc[3] = __builtin_amdgcn_wmma_f32_16x16x32_f16(
                false, A0, false, B3, (short)0, acc[3], false, false);
            acc[4] = __builtin_amdgcn_wmma_f32_16x16x32_f16(
                false, A1, false, B0, (short)0, acc[4], false, false);
            acc[5] = __builtin_amdgcn_wmma_f32_16x16x32_f16(
                false, A1, false, B1, (short)0, acc[5], false, false);
            acc[6] = __builtin_amdgcn_wmma_f32_16x16x32_f16(
                false, A1, false, B2, (short)0, acc[6], false, false);
            acc[7] = __builtin_amdgcn_wmma_f32_16x16x32_f16(
                false, A1, false, B3, (short)0, acc[7], false, false);
        }

        // ---- fused epilogue: e = exp(2*dot - 2), diagonal masked out ----
        const int cTile = c0 + ct * BN;
#pragma unroll
        for (int r = 0; r < 2; ++r) {
            const int rowb = rBase + wave * 32 + r * 16 + lh * 8;
#pragma unroll
            for (int t = 0; t < 4; ++t) {
                const int col = cTile + t * 16 + l15;
#pragma unroll
                for (int v = 0; v < 8; ++v) {
                    float e = __expf(acc[r * 4 + t][v] * TAU_INV - SHIFT);
                    if (rowb + v == col) e = 0.f;
                    sumacc[r][v] += e;
                }
            }
        }
    }

    // Reduce partial row-sums across the 16-lane column groups, then
    // combine the 4 column chunks via f32 global atomics.
#pragma unroll
    for (int r = 0; r < 2; ++r)
#pragma unroll
        for (int v = 0; v < 8; ++v) {
#pragma unroll
            for (int m = 1; m < 16; m <<= 1)
                sumacc[r][v] += __shfl_xor(sumacc[r][v], m, 32);
        }
    if (l15 == 0) {
#pragma unroll
        for (int r = 0; r < 2; ++r) {
            const int rowb = rBase + wave * 32 + r * 16 + lh * 8;
#pragma unroll
            for (int v = 0; v < 8; ++v)
                atomicAdd(&S[rowb + v], sumacc[r][v]);
        }
    }
}

// ------------------------------------------------------------- positives ---
__global__ void __launch_bounds__(256)
ntx_pos(const _Float16* __restrict__ y, float* __restrict__ pos) {
    const int row  = blockIdx.x * 8 + (threadIdx.x >> 5);
    const int lane = threadIdx.x & 31;
    const int pair = row ^ HALFN;   // (row + bs) % N for power-of-two bs
    const _Float16* a = y + (size_t)row  * DIMK;
    const _Float16* b = y + (size_t)pair * DIMK;
    float s = 0.f;
#pragma unroll
    for (int i = 0; i < DIMK / 32; ++i) {
        const int k = lane + 32 * i;
        s += (float)a[k] * (float)b[k];
    }
#pragma unroll
    for (int m = 1; m < 32; m <<= 1) s += __shfl_xor(s, m, 32);
    if (lane == 0) pos[row] = s * TAU_INV;
}

// -------------------------------------------------------------- finalize ---
__global__ void __launch_bounds__(256)
ntx_fin(const float* __restrict__ S, const float* __restrict__ pos,
        float* __restrict__ out) {
    __shared__ float red[256];
    float s = 0.f;
    for (int i = threadIdx.x; i < NROWS; i += 256)
        s += SHIFT + __logf(S[i]) - pos[i];
    red[threadIdx.x] = s;
    __syncthreads();
    for (int o = 128; o > 0; o >>= 1) {
        if ((int)threadIdx.x < o) red[threadIdx.x] += red[threadIdx.x + o];
        __syncthreads();
    }
    if (threadIdx.x == 0) out[0] = red[0] / (float)NROWS;
}

// ----------------------------------------------------------------- launch ---
extern "C" void kernel_launch(void* const* d_in, const int* in_sizes, int n_in,
                              void* d_out, int out_size, void* d_ws, size_t ws_size,
                              hipStream_t stream) {
    (void)in_sizes; (void)n_in; (void)out_size; (void)ws_size;
    const float* xi = (const float*)d_in[0];
    const float* xj = (const float*)d_in[1];
    float* out = (float*)d_out;

    char* ws = (char*)d_ws;
    const size_t ybytes = (size_t)NROWS * DIMK * sizeof(_Float16); // 12.6 MB
    _Float16* y  = (_Float16*)ws;
    float*    S  = (float*)(ws + ybytes);                           // 32 KB
    float*    ps = (float*)(ws + ybytes + (size_t)NROWS * sizeof(float));

    // 1) normalize + zero S  (8192 waves)
    ntx_nrm<<<NROWS / 8, 256, 0, stream>>>(xi, xj, y, S);
    // 2) LDS-blocked WMMA similarity GEMM with fused shifted logsumexp
    dim3 grid(NROWS / BM, NROWS / CB);   // (64, 4)
    ntx_simlse<<<grid, 128, SMEM_BYTES, stream>>>(y, S);
    // 3) positive-pair logits
    ntx_pos<<<NROWS / 8, 256, 0, stream>>>(y, ps);
    // 4) final reduction -> scalar loss
    ntx_fin<<<1, 256, 0, stream>>>(S, ps, out);
}
